// GeometryAwareSelfAttentionBlock_63728724738250
// MI455X (gfx1250) — compile-verified, matrix operand order checked by reading, not run on hardware
//
#include <hip/hip_runtime.h>

// ---------------- problem constants (match reference setup_inputs) ----------
#define DMODEL 384
#define NHEAD  6
#define DHEAD  64
#define KNN    8
#define BATCH  2
#define NPT    4096
#define ROWS   (BATCH * NPT)

typedef __attribute__((ext_vector_type(16))) __bf16 v16bf;
typedef __attribute__((ext_vector_type(8)))  __bf16 v8bf;
typedef __attribute__((ext_vector_type(8)))  float  v8f;

#define V8F_ZERO {0.f,0.f,0.f,0.f,0.f,0.f,0.f,0.f}

__device__ __forceinline__ int laneid() { return threadIdx.x & 31; }

// ---------------------------------------------------------------------------
// WMMA 16x16x32 bf16 fragment loader (wave32, gfx1250 layout):
//   A-frag from row-major [R x K], or B-frag from K-transposed row-major [N x K]:
//     lanes 0-15 : row = r0+lane,    K in {k0..k0+7, k0+16..k0+23}
//     lanes 16-31: row = r0+lane-16, K in {k0+8..k0+15, k0+24..k0+31}
// Two contiguous 16B loads per lane.
__device__ __forceinline__ v16bf ldfrag(const __bf16* p, int ld, int r0, int k0) {
  int L = laneid();
  int r = r0 + (L & 15);
  int kb = k0 + ((L >> 4) << 3);
  const __bf16* base = p + (size_t)r * ld + kb;
  v8bf lo = *reinterpret_cast<const v8bf*>(base);
  v8bf hi = *reinterpret_cast<const v8bf*>(base + 16);
  v16bf a;
#pragma unroll
  for (int i = 0; i < 8; ++i) { a[i] = lo[i]; a[8 + i] = hi[i]; }
  return a;
}

// ---------------------------------------------------------------------------
// Weight convert + transpose: wT[n*K + k] = (bf16) w[k*N + n]
__global__ __launch_bounds__(256)
void wt_cvt(const float* __restrict__ w, __bf16* __restrict__ wt, int Kdim, int Ndim) {
  int i = blockIdx.x * blockDim.x + threadIdx.x;
  if (i >= Kdim * Ndim) return;
  int k = i / Ndim, n = i - k * Ndim;
  wt[(size_t)n * Kdim + k] = (__bf16)w[i];
}

// ---------------------------------------------------------------------------
// V transpose: vT[((b*NHEAD+h)*DHEAD+d)*NPT + n] = qkv[(b*NPT+n)*1152 + 768 + h*64 + d]
// n fastest -> coalesced writes; reads hit L2 (qkv just produced).
__global__ __launch_bounds__(256)
void v_transpose(const __bf16* __restrict__ qkv, __bf16* __restrict__ vT) {
  int i = blockIdx.x * blockDim.x + threadIdx.x;
  if (i >= BATCH * NHEAD * DHEAD * NPT) return;
  int n = i & (NPT - 1);
  int rest = i >> 12;            // /NPT
  int d = rest & (DHEAD - 1);
  rest >>= 6;                    // /DHEAD
  int h = rest % NHEAD;
  int b = rest / NHEAD;
  vT[i] = qkv[(size_t)(b * NPT + n) * (3 * DMODEL) + 2 * DMODEL + h * DHEAD + d];
}

// ---------------------------------------------------------------------------
// Row LayerNorm: fp32 in -> bf16 out.  One wave per row (D=384).
__global__ __launch_bounds__(256)
void ln_rows(const float* __restrict__ x, const float* __restrict__ g,
             const float* __restrict__ b, __bf16* __restrict__ y, int rows) {
  int row = blockIdx.x * (blockDim.x >> 5) + (threadIdx.x >> 5);
  if (row >= rows) return;
  int L = laneid();
  const float* xr = x + (size_t)row * DMODEL;
  float s = 0.f, ss = 0.f;
  for (int c = L; c < DMODEL; c += 32) { float v = xr[c]; s += v; ss += v * v; }
#pragma unroll
  for (int off = 16; off; off >>= 1) { s += __shfl_xor(s, off, 32); ss += __shfl_xor(ss, off, 32); }
  float mean = s * (1.0f / DMODEL);
  float var  = ss * (1.0f / DMODEL) - mean * mean;
  float rstd = rsqrtf(var + 1e-5f);
  __bf16* yr = y + (size_t)row * DMODEL;
  for (int c = L; c < DMODEL; c += 32)
    yr[c] = (__bf16)((xr[c] - mean) * rstd * g[c] + b[c]);
}

// ---------------------------------------------------------------------------
// Generic WMMA GEMM, 32x32 C block per wave (2x2 tiles -> 4 frag loads per
// 4 WMMAs).  A bf16 row-major (lda), BT bf16 row-major [N,K] (ldb = K).
// M, N multiples of 32; K multiple of 32.
template<bool GELU, typename OutT>
__global__ __launch_bounds__(256)
void gemm_bf16_wmma(const __bf16* __restrict__ A, int lda,
                    const __bf16* __restrict__ BT, int ldb,
                    const float* __restrict__ bias,
                    const float* __restrict__ resid, int ldr,
                    OutT* __restrict__ C, int ldc,
                    int M, int N, int Ktot) {
  int wave = blockIdx.x * (blockDim.x >> 5) + (threadIdx.x >> 5);
  int blocksN = N >> 5;
  int blocks = (M >> 5) * blocksN;
  if (wave >= blocks) return;
  int bm = wave / blocksN, bn = wave - bm * blocksN;
  int L = laneid();
  int hi = L >> 4, ln = L & 15;
  v8f acc[2][2] = {{V8F_ZERO, V8F_ZERO}, {V8F_ZERO, V8F_ZERO}};
  for (int k = 0; k < Ktot; k += 32) {
    if (k + 32 < Ktot) {  // speculative prefetch of next K-slab
      __builtin_prefetch(A + (size_t)(bm * 32 + ln) * lda + k + 32, 0, 1);
      __builtin_prefetch(BT + (size_t)(bn * 32 + ln) * ldb + k + 32, 0, 1);
    }
    v16bf a0 = ldfrag(A, lda, bm * 32, k);
    v16bf a1 = ldfrag(A, lda, bm * 32 + 16, k);
    v16bf b0 = ldfrag(BT, ldb, bn * 32, k);
    v16bf b1 = ldfrag(BT, ldb, bn * 32 + 16, k);
    acc[0][0] = __builtin_amdgcn_wmma_f32_16x16x32_bf16(false, a0, false, b0, (short)0, acc[0][0], false, false);
    acc[0][1] = __builtin_amdgcn_wmma_f32_16x16x32_bf16(false, a0, false, b1, (short)0, acc[0][1], false, false);
    acc[1][0] = __builtin_amdgcn_wmma_f32_16x16x32_bf16(false, a1, false, b0, (short)0, acc[1][0], false, false);
    acc[1][1] = __builtin_amdgcn_wmma_f32_16x16x32_bf16(false, a1, false, b1, (short)0, acc[1][1], false, false);
  }
#pragma unroll
  for (int i = 0; i < 2; ++i) {
#pragma unroll
    for (int j = 0; j < 2; ++j) {
      int col = bn * 32 + j * 16 + ln;
      int rbase = bm * 32 + i * 16 + 8 * hi;
      float bv = bias ? bias[col] : 0.0f;
#pragma unroll
      for (int r = 0; r < 8; ++r) {
        int row = rbase + r;
        float x = acc[i][j][r] + bv;
        if (GELU) {
          float u = 0.7978845608028654f * (x + 0.044715f * x * x * x);
          x = 0.5f * x * (1.0f + tanhf(u));
        }
        if (resid) x += resid[(size_t)row * ldr + col];
        C[(size_t)row * ldc + col] = (OutT)x;
      }
    }
  }
}

// ---------------------------------------------------------------------------
// Flash attention.  qkv bf16 [ROWS x 1152] supplies Q,K; vT bf16
// [B][H][DHEAD][NPT] supplies V pre-transposed so every fragment load is two
// contiguous 16B loads.  One wave per (b, h, 16-query tile); 64-key tiles.
// Online softmax: row max via one 4-step shuffle reduce per row per 64 keys;
// row sums computed with an extra WMMA against an all-ones B fragment (sum
// lands in the same C-layout as the O accumulators), eliminating sum shuffles.
__global__ __launch_bounds__(256)
void flash_attn(const __bf16* __restrict__ qkv, const __bf16* __restrict__ vT,
                __bf16* __restrict__ outO) {
  __shared__ __bf16 plds[8 * 16 * 64];
  const int ld = 3 * DMODEL;
  const int qtiles = NPT / 16;
  int wid = blockIdx.x * (blockDim.x >> 5) + (threadIdx.x >> 5);
  if (wid >= BATCH * NHEAD * qtiles) return;
  int qt = wid % qtiles;
  int h  = (wid / qtiles) % NHEAD;
  int b  = wid / (qtiles * NHEAD);
  int L = laneid();
  int hi = L >> 4, ln = L & 15;
  size_t rowbase = (size_t)b * NPT;
  const __bf16* Q  = qkv + rowbase * ld + h * DHEAD;
  const __bf16* Kp = qkv + rowbase * ld + DMODEL + h * DHEAD;
  const __bf16* Vt = vT + (size_t)(b * NHEAD + h) * DHEAD * NPT;  // [64][NPT]
  __bf16* P = plds + (threadIdx.x >> 5) * (16 * 64);

  v16bf qf0 = ldfrag(Q, ld, qt * 16, 0);
  v16bf qf1 = ldfrag(Q, ld, qt * 16, 32);
  v16bf onesb;
#pragma unroll
  for (int i = 0; i < 16; ++i) onesb[i] = (__bf16)1.0f;

  v8f o0 = V8F_ZERO, o1 = V8F_ZERO, o2 = V8F_ZERO, o3 = V8F_ZERO;
  float mstat[8], lstat[8];
#pragma unroll
  for (int r = 0; r < 8; ++r) { mstat[r] = -1e30f; lstat[r] = 0.f; }
  const float scale = 0.125f;  // 1/sqrt(64)

  for (int k0 = 0; k0 < NPT; k0 += 64) {
    // S = Q(16x64) @ K^T(64x64): four 16x16 tiles, 2 K-steps each
    v8f s0 = V8F_ZERO, s1 = V8F_ZERO, s2 = V8F_ZERO, s3 = V8F_ZERO;
    s0 = __builtin_amdgcn_wmma_f32_16x16x32_bf16(false, qf0, false, ldfrag(Kp, ld, k0, 0),       (short)0, s0, false, false);
    s0 = __builtin_amdgcn_wmma_f32_16x16x32_bf16(false, qf1, false, ldfrag(Kp, ld, k0, 32),      (short)0, s0, false, false);
    s1 = __builtin_amdgcn_wmma_f32_16x16x32_bf16(false, qf0, false, ldfrag(Kp, ld, k0 + 16, 0),  (short)0, s1, false, false);
    s1 = __builtin_amdgcn_wmma_f32_16x16x32_bf16(false, qf1, false, ldfrag(Kp, ld, k0 + 16, 32), (short)0, s1, false, false);
    s2 = __builtin_amdgcn_wmma_f32_16x16x32_bf16(false, qf0, false, ldfrag(Kp, ld, k0 + 32, 0),  (short)0, s2, false, false);
    s2 = __builtin_amdgcn_wmma_f32_16x16x32_bf16(false, qf1, false, ldfrag(Kp, ld, k0 + 32, 32), (short)0, s2, false, false);
    s3 = __builtin_amdgcn_wmma_f32_16x16x32_bf16(false, qf0, false, ldfrag(Kp, ld, k0 + 48, 0),  (short)0, s3, false, false);
    s3 = __builtin_amdgcn_wmma_f32_16x16x32_bf16(false, qf1, false, ldfrag(Kp, ld, k0 + 48, 32), (short)0, s3, false, false);

    // row max: local max over 4 tiles, then one 4-step half-wave reduction/row
    float escale[8];
#pragma unroll
    for (int r = 0; r < 8; ++r) {
      s0[r] *= scale; s1[r] *= scale; s2[r] *= scale; s3[r] *= scale;
      float mx = fmaxf(fmaxf(s0[r], s1[r]), fmaxf(s2[r], s3[r]));
#pragma unroll
      for (int off = 8; off; off >>= 1) mx = fmaxf(mx, __shfl_xor(mx, off, 32));
      float mn = fmaxf(mstat[r], mx);
      escale[r] = __expf(mstat[r] - mn);
      mstat[r] = mn;
      s0[r] = __expf(s0[r] - mn);
      s1[r] = __expf(s1[r] - mn);
      s2[r] = __expf(s2[r] - mn);
      s3[r] = __expf(s3[r] - mn);
    }
    // C-layout -> LDS [16 rows][64 keys] as bf16
#pragma unroll
    for (int r = 0; r < 8; ++r) {
      __bf16* pr = P + (r + 8 * hi) * 64;
      pr[ln]      = (__bf16)s0[r];
      pr[16 + ln] = (__bf16)s1[r];
      pr[32 + ln] = (__bf16)s2[r];
      pr[48 + ln] = (__bf16)s3[r];
    }
    // LDS -> WMMA A layout: P as two 16x32 A fragments
    v16bf pa0, pa1;
    {
      const __bf16* pr = P + (size_t)ln * 64 + hi * 8;
      v8bf l0 = *reinterpret_cast<const v8bf*>(pr);
      v8bf h0 = *reinterpret_cast<const v8bf*>(pr + 16);
      v8bf l1 = *reinterpret_cast<const v8bf*>(pr + 32);
      v8bf h1 = *reinterpret_cast<const v8bf*>(pr + 48);
#pragma unroll
      for (int i = 0; i < 8; ++i) {
        pa0[i] = l0[i]; pa0[8 + i] = h0[i];
        pa1[i] = l1[i]; pa1[8 + i] = h1[i];
      }
    }
    // row sums of P via WMMA against all-ones B (replicated across columns)
    v8f ps = V8F_ZERO;
    ps = __builtin_amdgcn_wmma_f32_16x16x32_bf16(false, pa0, false, onesb, (short)0, ps, false, false);
    ps = __builtin_amdgcn_wmma_f32_16x16x32_bf16(false, pa1, false, onesb, (short)0, ps, false, false);
    // rescale running stats and O, then O += P(16x64) @ V(64x64)
#pragma unroll
    for (int r = 0; r < 8; ++r) {
      lstat[r] = lstat[r] * escale[r] + ps[r];
      o0[r] *= escale[r]; o1[r] *= escale[r]; o2[r] *= escale[r]; o3[r] *= escale[r];
    }
    o0 = __builtin_amdgcn_wmma_f32_16x16x32_bf16(false, pa0, false, ldfrag(Vt, NPT, 0,  k0),      (short)0, o0, false, false);
    o0 = __builtin_amdgcn_wmma_f32_16x16x32_bf16(false, pa1, false, ldfrag(Vt, NPT, 0,  k0 + 32), (short)0, o0, false, false);
    o1 = __builtin_amdgcn_wmma_f32_16x16x32_bf16(false, pa0, false, ldfrag(Vt, NPT, 16, k0),      (short)0, o1, false, false);
    o1 = __builtin_amdgcn_wmma_f32_16x16x32_bf16(false, pa1, false, ldfrag(Vt, NPT, 16, k0 + 32), (short)0, o1, false, false);
    o2 = __builtin_amdgcn_wmma_f32_16x16x32_bf16(false, pa0, false, ldfrag(Vt, NPT, 32, k0),      (short)0, o2, false, false);
    o2 = __builtin_amdgcn_wmma_f32_16x16x32_bf16(false, pa1, false, ldfrag(Vt, NPT, 32, k0 + 32), (short)0, o2, false, false);
    o3 = __builtin_amdgcn_wmma_f32_16x16x32_bf16(false, pa0, false, ldfrag(Vt, NPT, 48, k0),      (short)0, o3, false, false);
    o3 = __builtin_amdgcn_wmma_f32_16x16x32_bf16(false, pa1, false, ldfrag(Vt, NPT, 48, k0 + 32), (short)0, o3, false, false);
  }
  // normalize and write to attnO [ROWS x 384] (bf16, pre-Wo)
#pragma unroll
  for (int r = 0; r < 8; ++r) {
    float inv = 1.0f / lstat[r];
    size_t row = rowbase + (size_t)qt * 16 + r + 8 * hi;
    __bf16* orow = outO + row * DMODEL + h * DHEAD;
    orow[ln]      = (__bf16)(o0[r] * inv);
    orow[16 + ln] = (__bf16)(o1[r] * inv);
    orow[32 + ln] = (__bf16)(o2[r] * inv);
    orow[48 + ln] = (__bf16)(o3[r] * inv);
  }
}

// ---------------------------------------------------------------------------
// Brute-force k=8 nearest neighbors per point.  One wave per query point.
__device__ __forceinline__ float sel8f(const float* a, int h) {
  float v = 1e30f;
#pragma unroll
  for (int i = 0; i < 8; ++i) if (h == i) v = a[i];
  return v;
}
__device__ __forceinline__ int sel8i(const int* a, int h) {
  int v = -1;
#pragma unroll
  for (int i = 0; i < 8; ++i) if (h == i) v = a[i];
  return v;
}

__global__ __launch_bounds__(256)
void knn_topk(const float* __restrict__ coords, int* __restrict__ idxout) {
  int wid = blockIdx.x * (blockDim.x >> 5) + (threadIdx.x >> 5);
  if (wid >= ROWS) return;
  int b = wid / NPT, n = wid - b * NPT;
  int L = laneid();
  const float* cb = coords + (size_t)b * NPT * 3;
  float qx = cb[n * 3], qy = cb[n * 3 + 1], qz = cb[n * 3 + 2];
  float d[8]; int id[8];
#pragma unroll
  for (int i = 0; i < 8; ++i) { d[i] = 1e30f; id[i] = -1; }
  for (int j = L; j < NPT; j += 32) {
    float dx = cb[j * 3] - qx, dy = cb[j * 3 + 1] - qy, dz = cb[j * 3 + 2] - qz;
    float dist = dx * dx + dy * dy + dz * dz;
    if (dist < d[7]) {
      float cd = dist; int ci = j;
#pragma unroll
      for (int i = 0; i < 8; ++i)
        if (cd < d[i]) { float td = d[i]; int ti = id[i]; d[i] = cd; id[i] = ci; cd = td; ci = ti; }
    }
  }
  // cross-lane merge: 8 rounds of wave-min over each lane's sorted head
  int hsel = 0, res[8];
#pragma unroll
  for (int r = 0; r < 8; ++r) {
    float cand = sel8f(d, hsel);
    int cid = sel8i(id, hsel);
    float best = cand; int bl = L;
#pragma unroll
    for (int off = 16; off; off >>= 1) {
      float od = __shfl_xor(best, off, 32);
      int ob = __shfl_xor(bl, off, 32);
      if (od < best || (od == best && ob < bl)) { best = od; bl = ob; }
    }
    res[r] = __shfl(cid, bl, 32);
    if (L == bl) hsel++;
  }
  if (L == 0) {
    int* o = idxout + (size_t)wid * KNN;
#pragma unroll
    for (int r = 0; r < 8; ++r) o[r] = res[r];
  }
}

// ---------------------------------------------------------------------------
// Graph gather attention: one wave per point.  q,kf,vf fp32 [ROWS x 384].
// Writes bf16 result into cat[:, 384:768] (ldcat = 768).
__global__ __launch_bounds__(256)
void graph_gather(const float* __restrict__ qg, const float* __restrict__ kg,
                  const float* __restrict__ vg, const int* __restrict__ idx,
                  __bf16* __restrict__ cat) {
  int wid = blockIdx.x * (blockDim.x >> 5) + (threadIdx.x >> 5);
  if (wid >= ROWS) return;
  int b = wid / NPT;
  int L = laneid();
  const float* qr = qg + (size_t)wid * DMODEL;
  const int* ix = idx + (size_t)wid * KNN;
  size_t base = (size_t)b * NPT;
  float s[8];
#pragma unroll
  for (int k = 0; k < KNN; ++k) {
    const float* kr = kg + (base + ix[k]) * DMODEL;
    float p = 0.f;
    for (int c = L; c < DMODEL; c += 32) p += qr[c] * kr[c];
#pragma unroll
    for (int off = 16; off; off >>= 1) p += __shfl_xor(p, off, 32);
    s[k] = p * 0.05103103630798287f;  // 1/sqrt(384)
  }
  float mx = s[0];
#pragma unroll
  for (int k = 1; k < KNN; ++k) mx = fmaxf(mx, s[k]);
  float den = 0.f, a[8];
#pragma unroll
  for (int k = 0; k < KNN; ++k) { a[k] = __expf(s[k] - mx); den += a[k]; }
  float inv = 1.0f / den;
  __bf16* orow = cat + (size_t)wid * (2 * DMODEL) + DMODEL;
  for (int c = L; c < DMODEL; c += 32) {
    float acc = 0.f;
#pragma unroll
    for (int k = 0; k < KNN; ++k) acc += a[k] * vg[(base + ix[k]) * DMODEL + c];
    orow[c] = (__bf16)(acc * inv);
  }
}

// ---------------------------------------------------------------------------
extern "C" void kernel_launch(void* const* d_in, const int* in_sizes, int n_in,
                              void* d_out, int out_size, void* d_ws, size_t ws_size,
                              hipStream_t stream) {
  const float* coords     = (const float*)d_in[0];
  const float* features   = (const float*)d_in[1];
  const float* ln1_g      = (const float*)d_in[2];
  const float* ln1_b      = (const float*)d_in[3];
  const float* w_qkv      = (const float*)d_in[4];
  const float* w_attn_out = (const float*)d_in[5];
  const float* b_attn_out = (const float*)d_in[6];
  const float* w_gq       = (const float*)d_in[7];
  const float* w_gk       = (const float*)d_in[8];
  const float* w_gv       = (const float*)d_in[9];
  const float* w_merge    = (const float*)d_in[10];
  const float* b_merge    = (const float*)d_in[11];
  const float* ln2_g      = (const float*)d_in[12];
  const float* ln2_b      = (const float*)d_in[13];
  const float* w_ff1      = (const float*)d_in[14];
  const float* b_ff1      = (const float*)d_in[15];
  const float* w_ff2      = (const float*)d_in[16];
  const float* b_ff2      = (const float*)d_in[17];
  float* out = (float*)d_out;

  char* ws = (char*)d_ws;
  size_t off = 0;
  auto alloc = [&](size_t bytes) -> void* {
    void* p = ws + off;
    off = (off + bytes + 255) & ~(size_t)255;
    return p;
  };
  __bf16* nf     = (__bf16*)alloc((size_t)ROWS * DMODEL * 2);
  __bf16* qkv    = (__bf16*)alloc((size_t)ROWS * 3 * DMODEL * 2);
  __bf16* vT     = (__bf16*)alloc((size_t)ROWS * DMODEL * 2);
  __bf16* attnO  = (__bf16*)alloc((size_t)ROWS * DMODEL * 2);
  __bf16* cat    = (__bf16*)alloc((size_t)ROWS * 2 * DMODEL * 2);
  float*  qg     = (float*)alloc((size_t)ROWS * DMODEL * 4);
  float*  kg     = (float*)alloc((size_t)ROWS * DMODEL * 4);
  float*  vg     = (float*)alloc((size_t)ROWS * DMODEL * 4);
  int*    idx    = (int*)alloc((size_t)ROWS * KNN * 4);
  float*  feat2  = (float*)alloc((size_t)ROWS * DMODEL * 4);
  __bf16* hbuf   = (__bf16*)alloc((size_t)ROWS * DMODEL * 2);
  __bf16* ffh    = (__bf16*)alloc((size_t)ROWS * 2 * DMODEL * 2);
  __bf16* wqkvT  = (__bf16*)alloc((size_t)DMODEL * 3 * DMODEL * 2);
  __bf16* woT    = (__bf16*)alloc((size_t)DMODEL * DMODEL * 2);
  __bf16* wgqT   = (__bf16*)alloc((size_t)DMODEL * DMODEL * 2);
  __bf16* wgkT   = (__bf16*)alloc((size_t)DMODEL * DMODEL * 2);
  __bf16* wgvT   = (__bf16*)alloc((size_t)DMODEL * DMODEL * 2);
  __bf16* wmT    = (__bf16*)alloc((size_t)2 * DMODEL * DMODEL * 2);
  __bf16* wf1T   = (__bf16*)alloc((size_t)DMODEL * 2 * DMODEL * 2);
  __bf16* wf2T   = (__bf16*)alloc((size_t)2 * DMODEL * DMODEL * 2);

  auto cvt = [&](const float* w, __bf16* wt, int K, int N) {
    int n = K * N;
    wt_cvt<<<(n + 255) / 256, 256, 0, stream>>>(w, wt, K, N);
  };
  cvt(w_qkv,      wqkvT, DMODEL,     3 * DMODEL);
  cvt(w_attn_out, woT,   DMODEL,     DMODEL);
  cvt(w_gq,       wgqT,  DMODEL,     DMODEL);
  cvt(w_gk,       wgkT,  DMODEL,     DMODEL);
  cvt(w_gv,       wgvT,  DMODEL,     DMODEL);
  cvt(w_merge,    wmT,   2 * DMODEL, DMODEL);
  cvt(w_ff1,      wf1T,  DMODEL,     2 * DMODEL);
  cvt(w_ff2,      wf2T,  2 * DMODEL, DMODEL);

  auto gemm = [&](const __bf16* A, int lda, const __bf16* BT, int ldb,
                  const float* bias, const float* resid, int ldr,
                  void* C, int ldc, int M, int N, int K, bool gelu, bool outbf) {
    int tiles = (M / 32) * (N / 32);
    int blocks = (tiles + 7) / 8;
    if (outbf) {
      if (gelu) gemm_bf16_wmma<true,  __bf16><<<blocks, 256, 0, stream>>>(A, lda, BT, ldb, bias, resid, ldr, (__bf16*)C, ldc, M, N, K);
      else      gemm_bf16_wmma<false, __bf16><<<blocks, 256, 0, stream>>>(A, lda, BT, ldb, bias, resid, ldr, (__bf16*)C, ldc, M, N, K);
    } else {
      if (gelu) gemm_bf16_wmma<true,  float><<<blocks, 256, 0, stream>>>(A, lda, BT, ldb, bias, resid, ldr, (float*)C, ldc, M, N, K);
      else      gemm_bf16_wmma<false, float><<<blocks, 256, 0, stream>>>(A, lda, BT, ldb, bias, resid, ldr, (float*)C, ldc, M, N, K);
    }
  };

  const int rowBlocks = (ROWS / 8);  // 8 waves per 256-thread block

  // 1) LN1 -> nf (bf16)
  ln_rows<<<rowBlocks, 256, 0, stream>>>(features, ln1_g, ln1_b, nf, ROWS);
  // 2) qkv = nf @ w_qkv  (bf16 out)
  gemm(nf, DMODEL, wqkvT, DMODEL, nullptr, nullptr, 0, qkv, 3 * DMODEL,
       ROWS, 3 * DMODEL, DMODEL, false, true);
  // 2b) pre-transpose V for contiguous WMMA B fragments
  v_transpose<<<(ROWS * DMODEL + 255) / 256, 256, 0, stream>>>(qkv, vT);
  // 3) flash MHA -> attnO (bf16)
  flash_attn<<<(BATCH * NHEAD * (NPT / 16)) / 8, 256, 0, stream>>>(qkv, vT, attnO);
  // 4) attn_feat = attnO @ w_attn_out + b -> cat[:, 0:384]
  gemm(attnO, DMODEL, woT, DMODEL, b_attn_out, nullptr, 0, cat, 2 * DMODEL,
       ROWS, DMODEL, DMODEL, false, true);
  // 5) graph q/k/v (fp32 out)
  gemm(nf, DMODEL, wgqT, DMODEL, nullptr, nullptr, 0, qg, DMODEL, ROWS, DMODEL, DMODEL, false, false);
  gemm(nf, DMODEL, wgkT, DMODEL, nullptr, nullptr, 0, kg, DMODEL, ROWS, DMODEL, DMODEL, false, false);
  gemm(nf, DMODEL, wgvT, DMODEL, nullptr, nullptr, 0, vg, DMODEL, ROWS, DMODEL, DMODEL, false, false);
  // 6) kNN indices
  knn_topk<<<rowBlocks, 256, 0, stream>>>(coords, idx);
  // 7) gather attention -> cat[:, 384:768]
  graph_gather<<<rowBlocks, 256, 0, stream>>>(qg, kg, vg, idx, cat);
  // 8) feat2 = features + cat @ w_merge + b_merge   (fp32)
  gemm(cat, 2 * DMODEL, wmT, 2 * DMODEL, b_merge, features, DMODEL, feat2, DMODEL,
       ROWS, DMODEL, 2 * DMODEL, false, false);
  // 9) LN2 -> h (bf16)
  ln_rows<<<rowBlocks, 256, 0, stream>>>(feat2, ln2_g, ln2_b, hbuf, ROWS);
  // 10) ffh = gelu(h @ w_ff1 + b_ff1) (bf16)
  gemm(hbuf, DMODEL, wf1T, DMODEL, b_ff1, nullptr, 0, ffh, 2 * DMODEL,
       ROWS, 2 * DMODEL, DMODEL, true, true);
  // 11) out = feat2 + ffh @ w_ff2 + b_ff2  (fp32)
  gemm(ffh, 2 * DMODEL, wf2T, 2 * DMODEL, b_ff2, feat2, DMODEL, out, DMODEL,
       ROWS, DMODEL, 2 * DMODEL, false, false);

  (void)n_in; (void)in_sizes; (void)out_size; (void)ws_size;
}
